// _LSTMClassifier_17394617549244
// MI455X (gfx1250) — matrix operand order repaired and checked by reading, not compiled, not run on hardware
//
#include <hip/hip_runtime.h>

// ---------------------------------------------------------------------------
// Fused LSTM classifier for gfx1250 (MI455X).
// bf16 WMMA, f32 accumulate. 3 waves cooperate per 16-row batch tile:
// wave jg owns gate columns 48*G + 16*jg + nn, so each lane holds i/f/g/o of
// the same hidden unit across its 4 accumulators (gate padding 36 -> 48).
// Weight B-fragments are loop-invariant and live entirely in VGPRs.
// h is exchanged through a double-buffered LDS A tile ([x_t | h | pad]).
// Gate nonlinearities use gfx1250's hardware v_tanh_f32 when available.
// Invalid-lane h values land in the A-tile pad columns (K=116..127) whose
// weight rows are zero, so the h store is branchless.
// ---------------------------------------------------------------------------

typedef __attribute__((ext_vector_type(16))) __bf16 v16bf;
typedef __attribute__((ext_vector_type(8)))  __bf16 v8bf;
typedef __attribute__((ext_vector_type(8)))  float  v8f;

#define BATCH   2048
#define LFULL   8000
#define TSTEPS  400      // 8000 / 20
#define INPUTD  80       // 4 * UNIT
#define HID     36
#define KPAD    128      // 80 (x) + 36 (h) + 12 pad = 4 WMMA K-steps of 32
#define MTILE   16       // batch rows per workgroup
#define WAVES   3        // wave jg = 0..2 handles hidden units 16*jg .. 16*jg+15
#define THREADS (WAVES * 32)

__device__ __forceinline__ __bf16 f2bf(float f) {
  unsigned u = __builtin_bit_cast(unsigned, f);
  u += 0x7FFFu + ((u >> 16) & 1u);            // round-to-nearest-even
  unsigned short s = (unsigned short)(u >> 16);
  return __builtin_bit_cast(__bf16, s);
}
__device__ __forceinline__ float bf2f(__bf16 b) {
  unsigned short s = __builtin_bit_cast(unsigned short, b);
  unsigned u = ((unsigned)s) << 16;
  return __builtin_bit_cast(float, u);
}

// --- fast transcendentals: prefer gfx1250 hardware v_tanh_f32 ---
#if __has_builtin(__builtin_amdgcn_tanhf)
__device__ __forceinline__ float tanhf_(float x) {
  return __builtin_amdgcn_tanhf(x);                       // v_tanh_f32
}
#else
__device__ __forceinline__ float tanhf_(float x) {
  const float e = __expf(2.f * x);                        // v_exp_f32
  return 1.f - 2.f * __builtin_amdgcn_rcpf(e + 1.f);      // v_rcp_f32
}
#endif
// sigmoid(g + b) = 0.5*tanh(0.5*g + 0.5*b) + 0.5 ; pass bh = 0.5*b.
__device__ __forceinline__ float sigmoid_fma(float g, float bh) {
  return fmaf(0.5f, tanhf_(fmaf(0.5f, g, bh)), 0.5f);
}

// 16-element bf16 fragment from two contiguous 8-element LDS chunks.
__device__ __forceinline__ v16bf ld_frag(const __bf16* p0, const __bf16* p1) {
  v8bf a = *(const v8bf*)p0;   // ds_load_b128
  v8bf b = *(const v8bf*)p1;   // ds_load_b128
  return __builtin_shufflevector(a, b, 0,1,2,3,4,5,6,7,8,9,10,11,12,13,14,15);
}

__global__ __launch_bounds__(THREADS) void lstm_fused_kernel(
    const float* __restrict__ x,
    const float* __restrict__ w_ih, const float* __restrict__ w_hh,
    const float* __restrict__ b_ih, const float* __restrict__ b_hh,
    const float* __restrict__ fc1_w, const float* __restrict__ fc1_b,
    const float* __restrict__ fc2_w, const float* __restrict__ fc2_b,
    float* __restrict__ out)
{
  // Double-buffered A tile: rows m = batch, cols K = [x_t (80) | h (36) | pad]
  __shared__ __bf16 sA[2][MTILE * KPAD];                // 2 * 4 KB

  const int tid  = threadIdx.x;
  const int jg   = tid >> 5;       // this wave's hidden-unit group (0..2)
  const int lane = tid & 31;
  const int half = lane >> 4;      // WMMA lane half
  const int nn   = lane & 15;

  // ---- zero both A buffers (h0 = 0, K-pad = 0) ----
  for (int idx = tid; idx < 2 * MTILE * KPAD; idx += THREADS)
    (&sA[0][0])[idx] = f2bf(0.f);

  // ---- loop-invariant weight B-fragments -> registers (16 x v16bf) ----
  // B k order chosen here (contiguous, same as A): lane (nn, half), elem e -> K = 32*ks + 16*half + e
  const int  j      = jg * 16 + nn;        // hidden unit this lane owns
  const bool jvalid = (j < HID);
  v16bf bfrag[4][4];
  float bias_g[4];                         // raw bias (tanh gate)
  float bias_h[4];                         // 0.5*bias (sigmoid gates)
  #pragma unroll
  for (int G = 0; G < 4; ++G) {
    const int row = G * HID + (jvalid ? j : 0);        // original gate row
    bias_g[G] = jvalid ? (b_ih[row] + b_hh[row]) : 0.f;
    bias_h[G] = 0.5f * bias_g[G];
    #pragma unroll
    for (int ks = 0; ks < 4; ++ks) {
      v16bf f;
      #pragma unroll
      for (int e = 0; e < 16; ++e) {
        const int K = ks * 32 + half * 16 + e;         // compile-time per e
        float v = 0.f;
        if (jvalid) {
          if (K < INPUTD)            v = w_ih[row * INPUTD + K];
          else if (K < INPUTD + HID) v = w_hh[row * HID + (K - INPUTD)];
        }
        f[e] = f2bf(v);                    // K >= 116 rows are always zero
      }
      bfrag[G][ks] = f;
    }
  }
  __syncthreads();

  const int b0 = blockIdx.x * MTILE;
  const float* xbase = x + (size_t)b0 * 4 * LFULL;

  // c state: creg[v] = c[row m = 8*half + v][unit j]
  float creg[8];
  #pragma unroll
  for (int v = 0; v < 8; ++v) creg[v] = 0.f;

  for (int t = 0; t < TSTEPS; ++t) {
    __bf16* aT = &sA[t & 1][0];          // holds x_t (staged now) + h_{t-1}
    __bf16* aN = &sA[(t + 1) & 1][0];    // receives h_t

    // ---- stage x_t (16 rows x 80 feats, bf16): 64 runs of 20 floats ----
    if (tid < 64) {
      const int m = tid >> 2, ch = tid & 3;
      const float* srcf = xbase + ((size_t)m * 4 + ch) * LFULL + t * 20;
      const float4* src = (const float4*)srcf;         // 16B aligned
      __bf16* dst = aT + m * KPAD + ch * 20;
      #pragma unroll
      for (int q = 0; q < 5; ++q) {
        const float4 v = src[q];                       // global_load_b128
        dst[q * 4 + 0] = f2bf(v.x);
        dst[q * 4 + 1] = f2bf(v.y);
        dst[q * 4 + 2] = f2bf(v.z);
        dst[q * 4 + 3] = f2bf(v.w);
      }
      if (t + 1 < TSTEPS) __builtin_prefetch(srcf + 20, 0, 0);
    }
    __syncthreads();

    // ---- A fragments (16-bit A layout): lane row m = nn ----
    v16bf afrag[4];
    #pragma unroll
    for (int ks = 0; ks < 4; ++ks) {
      const __bf16* p = aT + nn * KPAD + ks * 32 + half * 16;
      afrag[ks] = ld_frag(p, p + 8);   // K = base..base+15
    }

    // ---- 4 gate GEMMs for this wave's unit group; C starts at inline 0 ----
    v8f g4[4];
    #pragma unroll
    for (int G = 0; G < 4; ++G) {
      v8f acc = {};                                    // inline-0 C operand
      #pragma unroll
      for (int ks = 0; ks < 4; ++ks)
        acc = __builtin_amdgcn_wmma_f32_16x16x32_bf16(
            false, afrag[ks], false, bfrag[G][ks], (short)0, acc, false, false);
      g4[G] = acc;
    }

    // ---- per-lane LSTM cell update; bias folded into one FMA per gate ----
    #pragma unroll
    for (int v = 0; v < 8; ++v) {
      const float ig = sigmoid_fma(g4[0][v], bias_h[0]);
      const float fg = sigmoid_fma(g4[1][v], bias_h[1]);
      const float gg = tanhf_(g4[2][v] + bias_g[2]);
      const float og = sigmoid_fma(g4[3][v], bias_h[3]);
      const float c  = fg * creg[v] + ig * gg;
      creg[v] = c;
      // Branchless: invalid lanes (j in 36..47) write K=116..127, whose weight
      // rows are zero in every B fragment -> no effect on any GEMM result.
      const int m = half * 8 + v;                      // D-layout row
      aN[m * KPAD + INPUTD + j] = f2bf(og * tanhf_(c));
    }
    __syncthreads();
  }

  // ---- head (wave 0 only): h_last is in sA[TSTEPS & 1] = sA[0] ----
  if (jg == 0) {
    const __bf16* hrow = &sA[TSTEPS & 1][0] + nn * KPAD + INPUTD;
    float z[16];
    #pragma unroll
    for (int jo = 0; jo < 16; ++jo) {
      float s = fc1_b[jo];
      #pragma unroll
      for (int k = 0; k < HID; ++k)
        s += bf2f(hrow[k]) * fc1_w[jo * HID + k];
      z[jo] = fmaxf(s, 0.f);
    }
    #pragma unroll
    for (int cc = 0; cc < 5; ++cc) {                   // lane half picks 5 classes
      const int cls = half * 5 + cc;
      float s = fc2_b[cls];
      #pragma unroll
      for (int jo = 0; jo < 16; ++jo) s += z[jo] * fc2_w[cls * 16 + jo];
      out[(size_t)(b0 + nn) * 10 + cls] = s;
    }
  }
}

extern "C" void kernel_launch(void* const* d_in, const int* in_sizes, int n_in,
                              void* d_out, int out_size, void* d_ws, size_t ws_size,
                              hipStream_t stream) {
  (void)in_sizes; (void)n_in; (void)d_ws; (void)ws_size; (void)out_size;
  const float* x     = (const float*)d_in[0];
  const float* w_ih  = (const float*)d_in[1];
  const float* w_hh  = (const float*)d_in[2];
  const float* b_ih  = (const float*)d_in[3];
  const float* b_hh  = (const float*)d_in[4];
  const float* fc1_w = (const float*)d_in[5];
  const float* fc1_b = (const float*)d_in[6];
  const float* fc2_w = (const float*)d_in[7];
  const float* fc2_b = (const float*)d_in[8];

  dim3 grid(BATCH / MTILE);   // 128 workgroups, 3 waves each = 384 waves
  dim3 block(THREADS);        // 96 threads
  hipLaunchKernelGGL(lstm_fused_kernel, grid, block, 0, stream,
                     x, w_ih, w_hh, b_ih, b_hh,
                     fc1_w, fc1_b, fc2_w, fc2_b, (float*)d_out);
}
